// Encoder_74088185856202
// MI455X (gfx1250) — matrix operand order repaired and checked by reading, not compiled, not run on hardware
//
#include <hip/hip_runtime.h>
#include <hip/hip_bf16.h>
#include <math.h>

// Problem constants (from reference)
#define B_ 16
#define L_ 1024
#define DM_ 512
#define H_ 8
#define DK_ 64
#define DV_ 64
#define DH_ 2048
#define EPS_ 1e-5f

typedef __attribute__((ext_vector_type(16))) __bf16 v16bf;
typedef __attribute__((ext_vector_type(8)))  float  v8f;
typedef unsigned int   u32;
typedef unsigned short u16;

union Frag16 { u32 d[8]; v16bf v; };

__device__ __forceinline__ u16 f2bf(float x) {
    union { __bf16 b; u16 u; } c; c.b = (__bf16)x; return c.u;
}

// ---------------------------------------------------------------------------
// f32 -> bf16 elementwise (packed pair per thread; n is even)
// ---------------------------------------------------------------------------
__global__ void enc_cvt_bf16(const float* __restrict__ in, u16* __restrict__ out, size_t n)
{
    size_t i = ((size_t)blockIdx.x * blockDim.x + threadIdx.x) * 2;
    if (i >= n) return;
    u32 p = (u32)f2bf(in[i]) | ((u32)f2bf(in[i + 1]) << 16);
    *(u32*)(out + i) = p;
}

// ---------------------------------------------------------------------------
// f32 (K x N) -> bf16 transposed (N x K), 32x32 LDS tile transpose
// ---------------------------------------------------------------------------
__global__ __launch_bounds__(256)
void enc_cvt_bf16_t(const float* __restrict__ in, u16* __restrict__ out, int K, int N)
{
    __shared__ u16 tile[32][33];
    const int k0 = blockIdx.y * 32, n0 = blockIdx.x * 32;
    const int c = threadIdx.x & 31, r8 = threadIdx.x >> 5;   // 8 rows per pass
#pragma unroll
    for (int p = 0; p < 4; ++p) {
        int r = r8 + p * 8;
        tile[r][c] = f2bf(in[(size_t)(k0 + r) * N + n0 + c]);
    }
    __syncthreads();
#pragma unroll
    for (int p = 0; p < 4; ++p) {
        int r = r8 + p * 8;
        out[(size_t)(n0 + r) * K + k0 + c] = tile[c][r];
    }
}

// ---------------------------------------------------------------------------
// GEMM: C[M,N] = A[M,K] @ B[K,N] + bias, A bf16 row-major, Bt = B^T bf16 (NxK).
// Block 256 thr = 8 waves; block tile 64x256; wave tile 32x64
// (8 x V_WMMA_F32_16X16X32_BF16 per k-step, B fragments reused 2x).
// LDS rows padded to 20 dwords: conflict-free (16 lanes x stride 20 -> 16
// distinct banks) AND 16B-aligned -> ds_load_b128 / ds_store_b128.
// Register-buffered pipeline over ping-pong LDS: tile k+1 global_load_b128s
// are issued (to 5 distinct uint4 regs, no wait) BEFORE the 8 WMMAs of tile k,
// and stored to the alternate LDS buffer AFTER compute -> global latency
// overlaps the WMMA stream. One barrier per k-step.
// If Cbf != null store bf16 (feeds next GEMM) else f32. M%64==N%256==K%32==0.
// ---------------------------------------------------------------------------
#define LB_S 20
#define LA_S 20

__global__ __launch_bounds__(256)
void enc_wmma_gemm(const u16* __restrict__ A, const u16* __restrict__ Bt,
                   const float* __restrict__ bias, float* __restrict__ C,
                   u16* __restrict__ Cbf, int M, int N, int K, int relu)
{
    __shared__ __align__(16) u32 lb[2][256 * LB_S];  // B tiles (20KB each)
    __shared__ __align__(16) u32 la[2][64 * LA_S];   // A tiles (5KB each)

    const int tid  = threadIdx.x;
    const int lane = tid & 31;
    const int wave = tid >> 5;
    const int wm   = wave >> 2;      // 0..1 : 32-row half
    const int wn   = wave & 3;       // 0..3 : 64-col strip
    const int mrow = lane & 15;
    const int hi   = lane >> 4;

    const int mb = blockIdx.y * 64;
    const int nb = blockIdx.x * 256;

    const uint4* A4 = (const uint4*)A;
    const uint4* B4 = (const uint4*)Bt;
    const int Kq = K >> 3;           // uint4 (8 bf16) per row
    const int nK = K >> 5;           // number of 32-wide k-steps

    // Per-thread staging coordinates (constant across k-steps)
    const int s_nc = tid >> 2, s_kk = tid & 3;        // B: 64-row groups x 4 chunks
    const int s_mr = tid >> 2;                        // A: 64 rows x 4 chunks

    v8f acc[2][4];
#pragma unroll
    for (int s = 0; s < 2; ++s)
#pragma unroll
        for (int t = 0; t < 4; ++t)
            acc[s][t] = (v8f){0.f,0.f,0.f,0.f,0.f,0.f,0.f,0.f};

    // Register staging buffers (distinct regs -> independent b128 loads)
    uint4 rb0, rb1, rb2, rb3, ra;

    auto stage_load = [&](int ks) {
        const size_t off = (size_t)ks * 4 + s_kk;
        rb0 = B4[(size_t)(nb + s_nc      ) * Kq + off];
        rb1 = B4[(size_t)(nb + s_nc +  64) * Kq + off];
        rb2 = B4[(size_t)(nb + s_nc + 128) * Kq + off];
        rb3 = B4[(size_t)(nb + s_nc + 192) * Kq + off];
        ra  = A4[(size_t)(mb + s_mr)       * Kq + off];
    };
    auto stage_store = [&](int bufi) {
        *(uint4*)&lb[bufi][(s_nc      ) * LB_S + s_kk * 4] = rb0;
        *(uint4*)&lb[bufi][(s_nc +  64) * LB_S + s_kk * 4] = rb1;
        *(uint4*)&lb[bufi][(s_nc + 128) * LB_S + s_kk * 4] = rb2;
        *(uint4*)&lb[bufi][(s_nc + 192) * LB_S + s_kk * 4] = rb3;
        *(uint4*)&la[bufi][s_mr * LA_S + s_kk * 4] = ra;
    };

    stage_load(0);
    stage_store(0);

    for (int ks = 0; ks < nK; ++ks) {
        __syncthreads();
        const int cur = ks & 1;
        const bool more = (ks + 1 < nK);
        if (more) stage_load(ks + 1);            // loads in flight during WMMAs
        if (ks + 2 < nK) {
            __builtin_prefetch(B4 + (size_t)(nb + tid) * Kq + (ks + 2) * 4, 0, 3);
            if (tid < 64)
                __builtin_prefetch(A4 + (size_t)(mb + tid) * Kq + (ks + 2) * 4, 0, 3);
        }

        // A fragments: lane<16 -> K{0..7,16..23}, lane>=16 -> K{8..15,24..31}
        Frag16 af[2];
#pragma unroll
        for (int s = 0; s < 2; ++s) {
            const u32* lap = &la[cur][(wm * 32 + s * 16 + mrow) * LA_S + hi * 4];
            uint4 lo = *(const uint4*)lap;
            uint4 hh = *(const uint4*)(lap + 8);
            af[s].d[0] = lo.x; af[s].d[1] = lo.y; af[s].d[2] = lo.z; af[s].d[3] = lo.w;
            af[s].d[4] = hh.x; af[s].d[5] = hh.y; af[s].d[6] = hh.z; af[s].d[7] = hh.w;
        }
#pragma unroll
        for (int t = 0; t < 4; ++t) {
            // B fragment: lane half hi covers K{hi*8..hi*8+7, 16+hi*8..16+hi*8+7}, col n=lane&15
            Frag16 fb;
            const u32* lbp = &lb[cur][(wn * 64 + t * 16 + mrow) * LB_S + hi * 4];
            uint4 q0 = *(const uint4*)lbp;
            uint4 q1 = *(const uint4*)(lbp + 8);
            fb.d[0] = q0.x; fb.d[1] = q0.y; fb.d[2] = q0.z; fb.d[3] = q0.w;
            fb.d[4] = q1.x; fb.d[5] = q1.y; fb.d[6] = q1.z; fb.d[7] = q1.w;

            acc[0][t] = __builtin_amdgcn_wmma_f32_16x16x32_bf16(
                false, af[0].v, false, fb.v, (short)0, acc[0][t], false, false);
            acc[1][t] = __builtin_amdgcn_wmma_f32_16x16x32_bf16(
                false, af[1].v, false, fb.v, (short)0, acc[1][t], false, false);
        }

        if (more) stage_store(cur ^ 1);          // waits for loads only here
    }

    // D layout: lane holds col n = lane&15; vgpr r holds row r + 8*hi
#pragma unroll
    for (int s = 0; s < 2; ++s) {
#pragma unroll
        for (int t = 0; t < 4; ++t) {
            const int n  = nb + wn * 64 + t * 16 + mrow;
            const float bv = bias ? bias[n] : 0.0f;
#pragma unroll
            for (int r = 0; r < 8; ++r) {
                const int m = mb + wm * 32 + s * 16 + r + 8 * hi;
                float v = acc[s][t][r] + bv;
                if (relu) v = v > 0.f ? v : 0.f;
                if (Cbf) Cbf[(size_t)m * N + n] = f2bf(v);
                else     C  [(size_t)m * N + n] = v;
            }
        }
    }
}

// ---------------------------------------------------------------------------
// M[b,h,l] = max_s(Q[l]·K[idx[l,s]]) - mean_s(...)  (Q,K: (B,L,H*DK) f32)
// ---------------------------------------------------------------------------
__global__ void enc_prob_m(const float* __restrict__ Q, const float* __restrict__ Kx,
                           const int* __restrict__ idx, float* __restrict__ Mout, int u)
{
    int t = blockIdx.x * blockDim.x + threadIdx.x;
    if (t >= B_ * H_ * L_) return;
    const int l = t & (L_ - 1);
    const int h = (t >> 10) & (H_ - 1);
    const int b = t >> 13;

    const float* q = Q + (size_t)(b * L_ + l) * (H_ * DK_) + h * DK_;
    float qreg[DK_];
#pragma unroll
    for (int j = 0; j < DK_; ++j) qreg[j] = q[j];

    float mx = -3.0e38f, sum = 0.f;
    for (int s = 0; s < u; ++s) {
        const int ks = idx[l * u + s];
        const float* krow = Kx + (size_t)(b * L_ + ks) * (H_ * DK_) + h * DK_;
        float d = 0.f;
#pragma unroll
        for (int j = 0; j < DK_; ++j) d += qreg[j] * krow[j];
        mx = fmaxf(mx, d);
        sum += d;
    }
    Mout[t] = mx - sum / (float)u;
}

// ---------------------------------------------------------------------------
// Top-u per (b,h) over L values (iterated argmax in LDS).
// ---------------------------------------------------------------------------
__global__ __launch_bounds__(256)
void enc_topk(const float* __restrict__ Mbuf, int* __restrict__ Mtop, int u)
{
    __shared__ float sval[L_];
    __shared__ float rval[256];
    __shared__ int   ridx[256];

    const int bh  = blockIdx.x;
    const int tid = threadIdx.x;
    for (int i = tid; i < L_; i += 256) sval[i] = Mbuf[(size_t)bh * L_ + i];
    __syncthreads();

    for (int it = 0; it < u; ++it) {
        float best = -3.0e38f; int bi = 0;
        for (int i = tid; i < L_; i += 256) {
            float v = sval[i];
            if (v > best) { best = v; bi = i; }
        }
        rval[tid] = best; ridx[tid] = bi;
        __syncthreads();
        for (int s = 128; s > 0; s >>= 1) {
            if (tid < s && rval[tid + s] > rval[tid]) {
                rval[tid] = rval[tid + s]; ridx[tid] = ridx[tid + s];
            }
            __syncthreads();
        }
        if (tid == 0) { Mtop[bh * u + it] = ridx[0]; sval[ridx[0]] = -3.0e38f; }
        __syncthreads();
    }
}

// vmean[b,h,d] = mean_l V[b,l,h,d]
__global__ void enc_vmean(const float* __restrict__ V, float* __restrict__ vmean)
{
    int t = blockIdx.x * blockDim.x + threadIdx.x;
    if (t >= B_ * H_ * DV_) return;
    const int d = t & (DV_ - 1);
    const int h = (t >> 6) & (H_ - 1);
    const int b = t >> 9;
    float s = 0.f;
    for (int l = 0; l < L_; ++l)
        s += V[(size_t)(b * L_ + l) * (H_ * DV_) + h * DV_ + d];
    vmean[t] = s * (1.0f / (float)L_);
}

// ctx default fill: ctx[b,l,h,d] = vmean[b,h,d]
__global__ void enc_fill_ctx(float* __restrict__ ctx, const float* __restrict__ vmean)
{
    size_t i = (size_t)blockIdx.x * blockDim.x + threadIdx.x;
    if (i >= (size_t)B_ * L_ * H_ * DV_) return;
    const int hd = (int)(i % (H_ * DV_));
    const int b  = (int)(i / ((size_t)L_ * H_ * DV_));
    ctx[i] = vmean[b * (H_ * DV_) + hd];
}

__global__ void enc_fill_const(float* __restrict__ p, size_t n, float v)
{
    size_t i = (size_t)blockIdx.x * blockDim.x + threadIdx.x;
    if (i < n) p[i] = v;
}

// ---------------------------------------------------------------------------
// Dense attention for the selected top-u query rows of each (b,h).
// ---------------------------------------------------------------------------
__global__ __launch_bounds__(256)
void enc_attn(const float* __restrict__ Q, const float* __restrict__ Kx,
              const float* __restrict__ V, const int* __restrict__ Mtop,
              float* __restrict__ ctx, float* __restrict__ score, int u)
{
    __shared__ float sc[L_];
    __shared__ float q[DK_];
    __shared__ float red[256];

    const int blk = blockIdx.x;                 // b*H*u + h*u + ui
    const int ui = blk % u;
    const int h  = (blk / u) % H_;
    const int b  = blk / (u * H_);
    const int tid = threadIdx.x;
    const int lsel = Mtop[(b * H_ + h) * u + ui];

    if (tid < DK_) q[tid] = Q[(size_t)(b * L_ + lsel) * (H_ * DK_) + h * DK_ + tid];
    __syncthreads();

    for (int l = tid; l < L_; l += 256) {
        const float* krow = Kx + (size_t)(b * L_ + l) * (H_ * DK_) + h * DK_;
        float d = 0.f;
#pragma unroll
        for (int j = 0; j < DK_; ++j) d += q[j] * krow[j];
        sc[l] = d * 0.125f;                      // 1/sqrt(64)
    }
    __syncthreads();

    float mx = -3.0e38f;
    for (int l = tid; l < L_; l += 256) mx = fmaxf(mx, sc[l]);
    red[tid] = mx; __syncthreads();
    for (int s = 128; s > 0; s >>= 1) {
        if (tid < s) red[tid] = fmaxf(red[tid], red[tid + s]);
        __syncthreads();
    }
    mx = red[0]; __syncthreads();

    float sum = 0.f;
    for (int l = tid; l < L_; l += 256) { float e = __expf(sc[l] - mx); sc[l] = e; sum += e; }
    red[tid] = sum; __syncthreads();
    for (int s = 128; s > 0; s >>= 1) {
        if (tid < s) red[tid] += red[tid + s];
        __syncthreads();
    }
    const float inv = 1.0f / red[0];
    __syncthreads();
    for (int l = tid; l < L_; l += 256) sc[l] *= inv;
    __syncthreads();

    float* srow = score + ((size_t)(b * H_ + h) * L_ + lsel) * L_;
    for (int l = tid; l < L_; l += 256) srow[l] = sc[l];

    if (tid < DV_) {
        float acc = 0.f;
        for (int l = 0; l < L_; ++l)
            acc += sc[l] * V[(size_t)(b * L_ + l) * (H_ * DV_) + h * DV_ + tid];
        ctx[(size_t)(b * L_ + lsel) * (H_ * DV_) + h * DV_ + tid] = acc;
    }
}

// ---------------------------------------------------------------------------
// out[row,:] = LayerNorm(a[row,:] + r[row,:]) * g + beta   (DM_=512, 256 thr)
// ---------------------------------------------------------------------------
__global__ __launch_bounds__(256)
void enc_add_ln(const float* __restrict__ a, const float* __restrict__ r,
                const float* __restrict__ g, const float* __restrict__ beta,
                float* __restrict__ out)
{
    __shared__ float red[256];
    const int row = blockIdx.x;
    const int tid = threadIdx.x;
    const float* ar = a + (size_t)row * DM_;
    const float* rr = r + (size_t)row * DM_;

    float v0 = ar[tid] + rr[tid];
    float v1 = ar[tid + 256] + rr[tid + 256];

    red[tid] = v0 + v1; __syncthreads();
    for (int s = 128; s > 0; s >>= 1) { if (tid < s) red[tid] += red[tid + s]; __syncthreads(); }
    const float mean = red[0] * (1.0f / (float)DM_);
    __syncthreads();

    const float d0 = v0 - mean, d1 = v1 - mean;
    red[tid] = d0 * d0 + d1 * d1; __syncthreads();
    for (int s = 128; s > 0; s >>= 1) { if (tid < s) red[tid] += red[tid + s]; __syncthreads(); }
    const float rstd = rsqrtf(red[0] * (1.0f / (float)DM_) + EPS_);

    float* orow = out + (size_t)row * DM_;
    orow[tid]       = g[tid]       * d0 * rstd + beta[tid];
    orow[tid + 256] = g[tid + 256] * d1 * rstd + beta[tid + 256];
}

// ---------------------------------------------------------------------------
extern "C" void kernel_launch(void* const* d_in, const int* in_sizes, int n_in,
                              void* d_out, int out_size, void* d_ws, size_t ws_size,
                              hipStream_t stream)
{
    const float* x     = (const float*)d_in[0];
    const float* Wq    = (const float*)d_in[1];
    const float* bq    = (const float*)d_in[2];
    const float* Wk    = (const float*)d_in[3];
    const float* bk    = (const float*)d_in[4];
    const float* Wv    = (const float*)d_in[5];
    const float* bv    = (const float*)d_in[6];
    const float* Wo    = (const float*)d_in[7];
    const float* bo    = (const float*)d_in[8];
    const float* g1    = (const float*)d_in[9];
    const float* beta1 = (const float*)d_in[10];
    const float* W1    = (const float*)d_in[11];
    const float* bf1   = (const float*)d_in[12];
    const float* W2    = (const float*)d_in[13];
    const float* bf2   = (const float*)d_in[14];
    const float* g2    = (const float*)d_in[15];
    const float* beta2 = (const float*)d_in[16];
    const int*   idx   = (const int*)d_in[17];
    const int u = in_sizes[17] / L_;   // 35

    float* x2    = (float*)d_out;                       // (B,L,DM)
    float* score = x2 + (size_t)B_ * L_ * DM_;          // (B,H,L,L)

    // Workspace layout (with lifetime-based overlays)
    const size_t NROW = (size_t)B_ * L_;                // 16384
    const size_t SZ   = NROW * DM_;                     // 8,388,608 floats
    char* base = (char*)d_ws;
    float* Q    = (float*)base;                         // R0
    float* Kb   = Q  + SZ;                              // R1
    float* V    = Kb + SZ;                              // R2
    float* ctx  = V  + SZ;                              // R3
    u16*   big  = (u16*)(ctx + SZ);                     // R4: NROW*DH_ bf16 (67MB)
    u16*   xbf     = big;                               // early life
    u16*   ffnh_bf = big;                               // late life
    u16*   Wq_t = big + NROW * DH_;
    u16*   Wk_t = Wq_t + (size_t)DM_ * DM_;
    u16*   Wv_t = Wk_t + (size_t)DM_ * DM_;
    u16*   Wo_t = Wv_t + (size_t)DM_ * DM_;
    u16*   W1_t = Wo_t + (size_t)DM_ * DM_;             // DH_ x DM_
    u16*   W2_t = W1_t + (size_t)DM_ * DH_;             // DM_ x DH_
    float* Mbuf = (float*)(W2_t + (size_t)DM_ * DH_);
    int*   Mtop = (int*)(Mbuf + (size_t)B_ * H_ * L_);
    float* vmean = (float*)(Mtop + B_ * H_ * u);
    // overlays (valid by lifetime):
    u16*   ctx_bf   = (u16*)Kb;   // after attn, K dead
    float* attn_out = Q;          // after attn, Q dead
    float* x1       = V;          // after attn, V dead
    u16*   x1_bf    = (u16*)ctx;  // after ctx->bf16 conversion, ctx f32 dead
    float* ffn_out  = Q;          // after LN1 consumed attn_out

    dim3 blk(256);
    auto gemm = [&](const u16* A, const u16* Bt, const float* bias,
                    float* C, u16* Cbf, int M, int N, int K, int relu) {
        dim3 grid(N / 256, M / 64);
        enc_wmma_gemm<<<grid, blk, 0, stream>>>(A, Bt, bias, C, Cbf, M, N, K, relu);
    };
    auto cvt = [&](const float* in, u16* out, size_t n) {
        enc_cvt_bf16<<<(unsigned)((n / 2 + 255) / 256), blk, 0, stream>>>(in, out, n);
    };
    auto cvt_t = [&](const float* in, u16* out, int K, int N) {
        dim3 grid(N / 32, K / 32);
        enc_cvt_bf16_t<<<grid, blk, 0, stream>>>(in, out, K, N);
    };

    // 0) one-time weight transpose+convert, activation convert
    cvt_t(Wq, Wq_t, DM_, DM_);
    cvt_t(Wk, Wk_t, DM_, DM_);
    cvt_t(Wv, Wv_t, DM_, DM_);
    cvt_t(Wo, Wo_t, DM_, DM_);
    cvt_t(W1, W1_t, DM_, DH_);
    cvt_t(W2, W2_t, DH_, DM_);
    cvt(x, xbf, NROW * DM_);

    // 1) QKV projections (WMMA GEMMs)
    gemm(xbf, Wq_t, bq, Q,  nullptr, (int)NROW, H_ * DK_, DM_, 0);
    gemm(xbf, Wk_t, bk, Kb, nullptr, (int)NROW, H_ * DK_, DM_, 0);
    gemm(xbf, Wv_t, bv, V,  nullptr, (int)NROW, H_ * DV_, DM_, 0);

    // 2) sparsity metric + top-u selection
    enc_prob_m<<<(B_ * H_ * L_ + 255) / 256, blk, 0, stream>>>(Q, Kb, idx, Mbuf, u);
    enc_topk<<<B_ * H_, blk, 0, stream>>>(Mbuf, Mtop, u);

    // 3) defaults: score = 1/L everywhere, ctx = broadcast V-mean
    {
        size_t nsc = (size_t)B_ * H_ * L_ * L_;
        enc_fill_const<<<(unsigned)((nsc + 255) / 256), blk, 0, stream>>>(score, nsc, 1.0f / (float)L_);
        enc_vmean<<<(B_ * H_ * DV_ + 255) / 256, blk, 0, stream>>>(V, vmean);
        size_t nctx = (size_t)B_ * L_ * H_ * DV_;
        enc_fill_ctx<<<(unsigned)((nctx + 255) / 256), blk, 0, stream>>>(ctx, vmean);
    }

    // 4) dense attention for selected rows (overwrites score rows + ctx rows)
    enc_attn<<<B_ * H_ * u, blk, 0, stream>>>(Q, Kb, V, Mtop, ctx, score, u);

    // 5) output projection + LN1
    cvt(ctx, ctx_bf, NROW * DM_);
    gemm(ctx_bf, Wo_t, bo, attn_out, nullptr, (int)NROW, DM_, H_ * DV_, 0);
    enc_add_ln<<<(int)NROW, blk, 0, stream>>>(attn_out, x, g1, beta1, x1);

    // 6) FFN: GEMM1 writes relu'd bf16 directly (no f32 hidden buffer), GEMM2, LN2
    cvt(x1, x1_bf, NROW * DM_);
    gemm(x1_bf, W1_t, bf1, nullptr, ffnh_bf, (int)NROW, DH_, DM_, 1);
    gemm(ffnh_bf, W2_t, bf2, ffn_out, nullptr, (int)NROW, DM_, DH_, 0);
    enc_add_ln<<<(int)NROW, blk, 0, stream>>>(ffn_out, x1, g2, beta2, x2);
}
